// DecoderLayer_86878598463633
// MI455X (gfx1250) — matrix operand order repaired and checked by reading, not compiled, not run on hardware
//
#include <hip/hip_runtime.h>

#define B_  16
#define S_  1024
#define T_  256
#define C_  512
#define H_  8
#define DH_ 64
#define BS_ (B_ * S_)   // 16384
#define BT_ (B_ * T_)   // 4096

typedef __attribute__((ext_vector_type(16))) __bf16 v16bf;
typedef __attribute__((ext_vector_type(8)))  float  v8f;

union AF {
  v16bf bf;
  uint4 q[2];
  unsigned u[8];
};

// round-to-nearest-even fp32 -> bf16 pair pack (lo, hi)
__device__ __forceinline__ unsigned pk2(float a, float b) {
  unsigned ua = __float_as_uint(a);
  unsigned ub = __float_as_uint(b);
  ua = (ua + 0x7fffu + ((ua >> 16) & 1u)) >> 16;
  ub = (ub + 0x7fffu + ((ub >> 16) & 1u)) >> 16;
  return ua | (ub << 16);
}
__device__ __forceinline__ unsigned short bf1(float a) {
  unsigned ua = __float_as_uint(a);
  return (unsigned short)((ua + 0x7fffu + ((ua >> 16) & 1u)) >> 16);
}
__device__ __forceinline__ float swishf(float x) {
  // v_rcp_f32 instead of IEEE divide sequence
  return x * __builtin_amdgcn_rcpf(1.0f + __expf(-x));
}
__device__ __forceinline__ v8f wmma_bf16(const AF& a, const AF& b, v8f c) {
  return __builtin_amdgcn_wmma_f32_16x16x32_bf16(false, a.bf, false, b.bf,
                                                 (short)0, c, false, false);
}
__device__ __forceinline__ float wred_sum32(float v) {
  v += __shfl_xor(v, 16, 32);
  v += __shfl_xor(v, 8, 32);
  v += __shfl_xor(v, 4, 32);
  v += __shfl_xor(v, 2, 32);
  v += __shfl_xor(v, 1, 32);
  return v;
}
__device__ __forceinline__ float wred_max32(float v) {
  v = fmaxf(v, __shfl_xor(v, 16, 32));
  v = fmaxf(v, __shfl_xor(v, 8, 32));
  v = fmaxf(v, __shfl_xor(v, 4, 32));
  v = fmaxf(v, __shfl_xor(v, 2, 32));
  v = fmaxf(v, __shfl_xor(v, 1, 32));
  return v;
}

// ---------------------------------------------------------------------------
// FiLM params: out[b,c] = sum_k sigma[b,k] * W[k,c] + bias[c]   (B*C threads)
// ---------------------------------------------------------------------------
__global__ __launch_bounds__(256) void film_kernel(
    const float* __restrict__ sigma, const float* __restrict__ W,
    const float* __restrict__ bias, float* __restrict__ out) {
  int idx = blockIdx.x * 256 + threadIdx.x;   // B_*C_ total
  int b = idx >> 9;                           // / C_
  int c = idx & (C_ - 1);
  const float* sp = sigma + b * C_;
  float a = 0.0f;
  for (int k = 0; k < C_; ++k) a += sp[k] * W[(size_t)k * C_ + c];
  out[idx] = a + bias[c];
}

// ---------------------------------------------------------------------------
// bf16 WMMA GEMM: out[M,N] = act( preact(A)[M,K] @ W[K,N] + bias ) + resid
// WG = 256 thr (8 waves), tile 128x128, K-step 32.
// Double-buffered LDS: next tile's global_load_b128s issued before the WMMA
// block of the current tile; one barrier per K-step; prefetch 2 tiles ahead.
// ---------------------------------------------------------------------------
__global__ __launch_bounds__(256) void gemm_bf16_kernel(
    const float* __restrict__ A, const float* __restrict__ W,
    const float* __restrict__ bias, const float* __restrict__ resid,
    float* __restrict__ out, int M, int N, int K, int preSwish, int postSwish) {
  __shared__ unsigned ldsA[2][128 * 16];   // [row][kpair]  (bf16 pairs along K)
  __shared__ unsigned ldsB[2][16 * 128];   // [kpair][n]    pair p -> k=2p,2p+1
  const int tid  = threadIdx.x;
  const int lane = tid & 31;
  const int wv   = tid >> 5;
  const int wrow = wv >> 2;             // 0..1  -> 64 rows each
  const int wcol = wv & 3;              // 0..3  -> 32 cols each
  const int gm = blockIdx.y * 128;
  const int gn = blockIdx.x * 128;

  v8f acc[4][2];
#pragma unroll
  for (int i = 0; i < 4; ++i)
#pragma unroll
    for (int j = 0; j < 2; ++j)
#pragma unroll
      for (int r = 0; r < 8; ++r) acc[i][j][r] = 0.0f;

  const int arow  = tid >> 1;
  const int aseg  = (tid & 1) * 16;
  const int bpair = tid >> 4;           // 0..15 -> k = 2*bpair
  const int bn    = (tid & 15) * 8;
  const int ml = lane & 15;
  const int hf = lane >> 4;

  const float* abase = A + (size_t)(gm + arow) * K + aseg;
  const float* bbase = W + (size_t)(2 * bpair) * N + (gn + bn);

  // register staging of the current global tile
  float fa[16];   // A: 16 contiguous K values of one row
  float fb[16];   // B: [0..7] = row k cols, [8..15] = row k+1 cols
  {
#pragma unroll
    for (int g = 0; g < 4; ++g) {
      float4 f = *(const float4*)(abase + g * 4);
      fa[g * 4 + 0] = f.x; fa[g * 4 + 1] = f.y;
      fa[g * 4 + 2] = f.z; fa[g * 4 + 3] = f.w;
    }
#pragma unroll
    for (int g = 0; g < 2; ++g) {
      float4 f = *(const float4*)(bbase + g * 4);
      fb[g * 4 + 0] = f.x; fb[g * 4 + 1] = f.y;
      fb[g * 4 + 2] = f.z; fb[g * 4 + 3] = f.w;
    }
#pragma unroll
    for (int g = 0; g < 2; ++g) {
      float4 f = *(const float4*)(bbase + N + g * 4);
      fb[8 + g * 4 + 0] = f.x; fb[8 + g * 4 + 1] = f.y;
      fb[8 + g * 4 + 2] = f.z; fb[8 + g * 4 + 3] = f.w;
    }
  }

  int buf = 0;
  for (int k0 = 0; k0 < K; k0 += 32) {
    // ---- convert + store the staged tile into lds[buf] ----
    {
      float ga[16];
#pragma unroll
      for (int e = 0; e < 16; ++e) ga[e] = fa[e];
      if (preSwish) {
#pragma unroll
        for (int e = 0; e < 16; ++e) ga[e] = swishf(ga[e]);
      }
      uint4 w0, w1;
      w0.x = pk2(ga[0], ga[1]);   w0.y = pk2(ga[2], ga[3]);
      w0.z = pk2(ga[4], ga[5]);   w0.w = pk2(ga[6], ga[7]);
      w1.x = pk2(ga[8], ga[9]);   w1.y = pk2(ga[10], ga[11]);
      w1.z = pk2(ga[12], ga[13]); w1.w = pk2(ga[14], ga[15]);
      unsigned* da = &ldsA[buf][arow * 16 + (aseg >> 1)];
      *(uint4*)(da + 0) = w0;
      *(uint4*)(da + 4) = w1;

      uint4 v0, v1;
      v0.x = pk2(fb[0], fb[8]);  v0.y = pk2(fb[1], fb[9]);
      v0.z = pk2(fb[2], fb[10]); v0.w = pk2(fb[3], fb[11]);
      v1.x = pk2(fb[4], fb[12]); v1.y = pk2(fb[5], fb[13]);
      v1.z = pk2(fb[6], fb[14]); v1.w = pk2(fb[7], fb[15]);
      unsigned* db = &ldsB[buf][bpair * 128 + bn];
      *(uint4*)(db + 0) = v0;
      *(uint4*)(db + 4) = v1;
    }
    __syncthreads();

    // ---- issue next tile's global loads (overlap with WMMAs below) ----
    if (k0 + 32 < K) {
      const float* ap = abase + (k0 + 32);
#pragma unroll
      for (int g = 0; g < 4; ++g) {
        float4 f = *(const float4*)(ap + g * 4);
        fa[g * 4 + 0] = f.x; fa[g * 4 + 1] = f.y;
        fa[g * 4 + 2] = f.z; fa[g * 4 + 3] = f.w;
      }
      const float* bp = bbase + (size_t)(k0 + 32) * N;
#pragma unroll
      for (int g = 0; g < 2; ++g) {
        float4 f = *(const float4*)(bp + g * 4);
        fb[g * 4 + 0] = f.x; fb[g * 4 + 1] = f.y;
        fb[g * 4 + 2] = f.z; fb[g * 4 + 3] = f.w;
      }
#pragma unroll
      for (int g = 0; g < 2; ++g) {
        float4 f = *(const float4*)(bp + N + g * 4);
        fb[8 + g * 4 + 0] = f.x; fb[8 + g * 4 + 1] = f.y;
        fb[8 + g * 4 + 2] = f.z; fb[8 + g * 4 + 3] = f.w;
      }
      if (k0 + 64 < K) {
        __builtin_prefetch(abase + (k0 + 64), 0, 1);
        __builtin_prefetch(bbase + (size_t)(k0 + 64) * N, 0, 1);
      }
    }

    // ---- compute from lds[buf] ----
    AF afr[4], bfr[2];
#pragma unroll
    for (int i = 0; i < 4; ++i) {
      const unsigned* ar = &ldsA[buf][(wrow * 64 + i * 16 + ml) * 16];
      afr[i].q[0] = *(const uint4*)(ar + hf * 8);       // k = hf*16 .. +7
      afr[i].q[1] = *(const uint4*)(ar + hf * 8 + 4);   // k = hf*16+8 .. +7
    }
#pragma unroll
    for (int j = 0; j < 2; ++j) {
      const int col = wcol * 32 + j * 16 + ml;
#pragma unroll
      for (int t = 0; t < 8; ++t) bfr[j].u[t] = ldsB[buf][(hf * 8 + t) * 128 + col];
    }
#pragma unroll
    for (int i = 0; i < 4; ++i)
#pragma unroll
      for (int j = 0; j < 2; ++j)
        acc[i][j] = wmma_bf16(afr[i], bfr[j], acc[i][j]);

    buf ^= 1;
  }

#pragma unroll
  for (int i = 0; i < 4; ++i) {
#pragma unroll
    for (int j = 0; j < 2; ++j) {
      const int col = gn + wcol * 32 + j * 16 + ml;
      const float bvv = bias ? bias[col] : 0.0f;
#pragma unroll
      for (int r = 0; r < 8; ++r) {
        const size_t row = (size_t)(gm + wrow * 64 + i * 16 + 8 * hf + r);
        float v = acc[i][j][r] + bvv;
        if (postSwish) v = swishf(v);
        if (resid) v += resid[row * N + col];
        out[row * N + col] = v;
      }
    }
  }
}

// ---------------------------------------------------------------------------
// LayerNorm + FiLM affine, wave-per-row (C=512, 16 elems/lane)
// out = (LN(in + pre)) * gamma[b] + beta[b] (+ post);  out2 = out + pe[t]
// ---------------------------------------------------------------------------
__global__ __launch_bounds__(256) void ln_affine_kernel(
    const float* __restrict__ in, const float* __restrict__ pre,
    const float* __restrict__ gamma, const float* __restrict__ beta,
    const float* __restrict__ post, const float* __restrict__ pe,
    float* __restrict__ out, float* __restrict__ out2, int L) {
  const int wv   = threadIdx.x >> 5;
  const int lane = threadIdx.x & 31;
  const size_t row = (size_t)blockIdx.x * 8 + wv;
  const int b = (int)(row / L);
  const int t = (int)(row % L);
  const float* rp = in + row * C_;
  float xv[16];
#pragma unroll
  for (int g = 0; g < 4; ++g) {
    float4 f = *(const float4*)(rp + lane * 16 + g * 4);
    xv[g * 4 + 0] = f.x; xv[g * 4 + 1] = f.y;
    xv[g * 4 + 2] = f.z; xv[g * 4 + 3] = f.w;
  }
  if (pre) {
    const float* pp = pre + row * C_;
#pragma unroll
    for (int g = 0; g < 4; ++g) {
      float4 f = *(const float4*)(pp + lane * 16 + g * 4);
      xv[g * 4 + 0] += f.x; xv[g * 4 + 1] += f.y;
      xv[g * 4 + 2] += f.z; xv[g * 4 + 3] += f.w;
    }
  }
  float s = 0.0f;
#pragma unroll
  for (int e = 0; e < 16; ++e) s += xv[e];
  const float mean = wred_sum32(s) * (1.0f / C_);
  float vs = 0.0f;
#pragma unroll
  for (int e = 0; e < 16; ++e) { float d = xv[e] - mean; vs += d * d; }
  const float inv = rsqrtf(wred_sum32(vs) * (1.0f / C_) + 1e-6f);
  const float* gp = gamma + (size_t)b * C_;
  const float* bp = beta + (size_t)b * C_;
#pragma unroll
  for (int e = 0; e < 16; ++e) {
    const int c = lane * 16 + e;
    float val = (xv[e] - mean) * inv * gp[c] + bp[c];
    if (post) val += post[row * C_ + c];
    out[row * C_ + c] = val;
    if (pe) out2[row * C_ + c] = val + pe[(size_t)t * C_ + c];
  }
}

// ---------------------------------------------------------------------------
// fp32 -> bf16 pair conversion (with scale); and V-transpose to [B,H,DH,L]
// ---------------------------------------------------------------------------
__global__ __launch_bounds__(256) void conv_bf16_kernel(
    const float* __restrict__ src, unsigned* __restrict__ dst, int npairs,
    float scale) {
  int i = blockIdx.x * 256 + threadIdx.x;
  if (i >= npairs) return;
  float2 f = *(const float2*)(src + 2 * (size_t)i);
  dst[i] = pk2(f.x * scale, f.y * scale);
}

__global__ __launch_bounds__(256) void convT_v_kernel(
    const float* __restrict__ v, unsigned* __restrict__ dst, int L) {
  int i = blockIdx.x * 256 + threadIdx.x;        // B*H*DH*(L/2) total
  const int Lh = L >> 1;
  int tp = i % Lh;
  int rem = i / Lh;
  int d = rem & 63; rem >>= 6;
  int h = rem & 7;
  int b = rem >> 3;
  const size_t base = (size_t)(b * L + 2 * tp) * C_ + h * DH_ + d;
  dst[i] = pk2(v[base], v[base + C_]);
}

// ---------------------------------------------------------------------------
// MHA1 scores: att_raw[b,h,q,t] = (q_scaled . k)  via 2 chained WMMAs (DH=64)
// one wave per (b, h, 16 q-rows); qb pre-scaled by 1/sqrt(DH)
// ---------------------------------------------------------------------------
__global__ __launch_bounds__(32) void mha1_scores_kernel(
    const unsigned* __restrict__ qb, const unsigned* __restrict__ kb,
    float* __restrict__ att) {
  const int blk = blockIdx.x;
  const int qt = blk & 63;
  const int h  = (blk >> 6) & 7;
  const int b  = blk >> 9;
  const int lane = threadIdx.x;
  const int ml = lane & 15;
  const int hf = lane >> 4;
  const unsigned* qrow = qb + (size_t)(b * S_ + qt * 16 + ml) * (C_ / 2) + h * 32;
  AF qa[2];
#pragma unroll
  for (int f = 0; f < 2; ++f) {
    qa[f].q[0] = *(const uint4*)(qrow + f * 16 + hf * 8);
    qa[f].q[1] = *(const uint4*)(qrow + f * 16 + hf * 8 + 4);
  }
  for (int t0 = 0; t0 < T_; t0 += 16) {
    const unsigned* kr = kb + (size_t)(b * T_ + t0 + ml) * (C_ / 2) + h * 32 + hf * 8;
    AF k0_, k1_;
    k0_.q[0] = *(const uint4*)(kr);      k0_.q[1] = *(const uint4*)(kr + 4);
    k1_.q[0] = *(const uint4*)(kr + 16); k1_.q[1] = *(const uint4*)(kr + 20);
    v8f c;
#pragma unroll
    for (int r = 0; r < 8; ++r) c[r] = 0.0f;
    c = wmma_bf16(qa[0], k0_, c);
    c = wmma_bf16(qa[1], k1_, c);
    float* orow = att + ((size_t)((b * H_ + h) * S_ + qt * 16 + 8 * hf)) * T_ + t0 + ml;
#pragma unroll
    for (int r = 0; r < 8; ++r) orow[(size_t)r * T_] = c[r];
  }
}

// wave-per-row softmax over T=256, normalize in place (this is `att` output)
__global__ __launch_bounds__(256) void softmax256_kernel(float* __restrict__ att) {
  const int wv   = threadIdx.x >> 5;
  const int lane = threadIdx.x & 31;
  float* p = att + ((size_t)blockIdx.x * 8 + wv) * T_;
  float4 a = *(float4*)(p + lane * 8);
  float4 c = *(float4*)(p + lane * 8 + 4);
  float mx = fmaxf(fmaxf(fmaxf(a.x, a.y), fmaxf(a.z, a.w)),
                   fmaxf(fmaxf(c.x, c.y), fmaxf(c.z, c.w)));
  mx = wred_max32(mx);
  a.x = __expf(a.x - mx); a.y = __expf(a.y - mx);
  a.z = __expf(a.z - mx); a.w = __expf(a.w - mx);
  c.x = __expf(c.x - mx); c.y = __expf(c.y - mx);
  c.z = __expf(c.z - mx); c.w = __expf(c.w - mx);
  float s = a.x + a.y + a.z + a.w + c.x + c.y + c.z + c.w;
  const float inv = __builtin_amdgcn_rcpf(wred_sum32(s));
  a.x *= inv; a.y *= inv; a.z *= inv; a.w *= inv;
  c.x *= inv; c.y *= inv; c.z *= inv; c.w *= inv;
  *(float4*)(p + lane * 8) = a;
  *(float4*)(p + lane * 8 + 4) = c;
}

// o[b,q,h*64+d] = sum_t att[b,h,q,t] * v[b,t,h*64+d]; K=T in 32-steps, WMMA
__global__ __launch_bounds__(32) void mha1_attv_kernel(
    const float* __restrict__ att, const unsigned* __restrict__ vT,
    float* __restrict__ o) {
  const int blk = blockIdx.x;
  const int qt = blk & 63;
  const int h  = (blk >> 6) & 7;
  const int b  = blk >> 9;
  const int lane = threadIdx.x;
  const int ml = lane & 15;
  const int hf = lane >> 4;
  const float* arow = att + (size_t)((b * H_ + h) * S_ + qt * 16 + ml) * T_;
  v8f acc[4];
#pragma unroll
  for (int j = 0; j < 4; ++j)
#pragma unroll
    for (int r = 0; r < 8; ++r) acc[j][r] = 0.0f;

  for (int t0 = 0; t0 < T_; t0 += 32) {
    AF af;
    const float* c0 = arow + t0 + hf * 16;
    const float* c1 = arow + t0 + hf * 16 + 8;
    float4 x0 = *(const float4*)(c0), x1 = *(const float4*)(c0 + 4);
    float4 y0 = *(const float4*)(c1), y1 = *(const float4*)(c1 + 4);
    af.u[0] = pk2(x0.x, x0.y); af.u[1] = pk2(x0.z, x0.w);
    af.u[2] = pk2(x1.x, x1.y); af.u[3] = pk2(x1.z, x1.w);
    af.u[4] = pk2(y0.x, y0.y); af.u[5] = pk2(y0.z, y0.w);
    af.u[6] = pk2(y1.x, y1.y); af.u[7] = pk2(y1.z, y1.w);
#pragma unroll
    for (int j = 0; j < 4; ++j) {
      const unsigned* vr = vT + ((size_t)((b * H_ + h) * DH_ + j * 16 + ml)) * (T_ / 2) +
                           (t0 >> 1) + hf * 8;
      AF vf;
      vf.q[0] = *(const uint4*)(vr);
      vf.q[1] = *(const uint4*)(vr + 4);
      acc[j] = wmma_bf16(af, vf, acc[j]);
    }
  }
#pragma unroll
  for (int j = 0; j < 4; ++j)
#pragma unroll
    for (int r = 0; r < 8; ++r)
      o[(size_t)(b * S_ + qt * 16 + 8 * hf + r) * C_ + h * DH_ + j * 16 + ml] = acc[j][r];
}

// ---------------------------------------------------------------------------
// MHA2 flash attention: one wave per (b,h,16 q-rows); streams S=1024 keys in
// 32-key steps with online softmax; never materializes the 512MB score tensor
// ---------------------------------------------------------------------------
__global__ __launch_bounds__(32) void flash_kernel(
    const unsigned* __restrict__ qb, const unsigned* __restrict__ kb,
    const unsigned* __restrict__ vT, float* __restrict__ o) {
  __shared__ __align__(16) unsigned short pl[16 * 32];  // P-tile relayout buffer
  const int blk = blockIdx.x;
  const int qt = blk & 63;
  const int h  = (blk >> 6) & 7;
  const int b  = blk >> 9;
  const int lane = threadIdx.x;
  const int ml = lane & 15;
  const int hf = lane >> 4;

  const unsigned* qrow = qb + (size_t)(b * S_ + qt * 16 + ml) * (C_ / 2) + h * 32;
  AF qa[2];
#pragma unroll
  for (int f = 0; f < 2; ++f) {
    qa[f].q[0] = *(const uint4*)(qrow + f * 16 + hf * 8);
    qa[f].q[1] = *(const uint4*)(qrow + f * 16 + hf * 8 + 4);
  }
  float mr[8], lr[8];
#pragma unroll
  for (int r = 0; r < 8; ++r) { mr[r] = -3.0e38f; lr[r] = 0.0f; }
  v8f acc[4];
#pragma unroll
  for (int j = 0; j < 4; ++j)
#pragma unroll
    for (int r = 0; r < 8; ++r) acc[j][r] = 0.0f;

  for (int t0 = 0; t0 < S_; t0 += 32) {
    const unsigned* kr0 = kb + (size_t)(b * S_ + t0 + ml) * (C_ / 2) + h * 32 + hf * 8;
    const unsigned* kr1 = kr0 + (size_t)16 * (C_ / 2);
    AF k00, k01, k10, k11;
    k00.q[0] = *(const uint4*)(kr0);      k00.q[1] = *(const uint4*)(kr0 + 4);
    k01.q[0] = *(const uint4*)(kr0 + 16); k01.q[1] = *(const uint4*)(kr0 + 20);
    k10.q[0] = *(const uint4*)(kr1);      k10.q[1] = *(const uint4*)(kr1 + 4);
    k11.q[0] = *(const uint4*)(kr1 + 16); k11.q[1] = *(const uint4*)(kr1 + 20);
    v8f s0, s1;
#pragma unroll
    for (int r = 0; r < 8; ++r) { s0[r] = 0.0f; s1[r] = 0.0f; }
    s0 = wmma_bf16(qa[0], k00, s0);
    s0 = wmma_bf16(qa[1], k01, s0);
    s1 = wmma_bf16(qa[0], k10, s1);
    s1 = wmma_bf16(qa[1], k11, s1);

    float p0[8], p1[8];
#pragma unroll
    for (int r = 0; r < 8; ++r) {
      float mx = fmaxf(s0[r], s1[r]);
      mx = fmaxf(mx, __shfl_xor(mx, 1, 32));
      mx = fmaxf(mx, __shfl_xor(mx, 2, 32));
      mx = fmaxf(mx, __shfl_xor(mx, 4, 32));
      mx = fmaxf(mx, __shfl_xor(mx, 8, 32));
      const float mnew = fmaxf(mr[r], mx);
      const float alpha = __expf(mr[r] - mnew);
      const float e0 = __expf(s0[r] - mnew);
      const float e1 = __expf(s1[r] - mnew);
      float rs = e0 + e1;
      rs += __shfl_xor(rs, 1, 32);
      rs += __shfl_xor(rs, 2, 32);
      rs += __shfl_xor(rs, 4, 32);
      rs += __shfl_xor(rs, 8, 32);
      lr[r] = lr[r] * alpha + rs;
      mr[r] = mnew;
      p0[r] = e0; p1[r] = e1;
#pragma unroll
      for (int j = 0; j < 4; ++j) acc[j][r] *= alpha;
    }
    // C/D-layout P -> A-fragment layout via tiny LDS round-trip
#pragma unroll
    for (int r = 0; r < 8; ++r) {
      pl[(r + 8 * hf) * 32 + ml]      = bf1(p0[r]);
      pl[(r + 8 * hf) * 32 + 16 + ml] = bf1(p1[r]);
    }
    __syncthreads();
    AF pf;
    pf.q[0] = *(const uint4*)&pl[ml * 32 + hf * 16];
    pf.q[1] = *(const uint4*)&pl[ml * 32 + hf * 16 + 8];
    __syncthreads();
#pragma unroll
    for (int j = 0; j < 4; ++j) {
      const unsigned* vr = vT + ((size_t)((b * H_ + h) * DH_ + j * 16 + ml)) * (S_ / 2) +
                           (t0 >> 1) + hf * 8;
      AF vf;
      vf.q[0] = *(const uint4*)(vr);
      vf.q[1] = *(const uint4*)(vr + 4);
      acc[j] = wmma_bf16(pf, vf, acc[j]);
    }
  }
#pragma unroll
  for (int r = 0; r < 8; ++r) {
    const float inv = __builtin_amdgcn_rcpf(lr[r]);
#pragma unroll
    for (int j = 0; j < 4; ++j)
      o[(size_t)(b * S_ + qt * 16 + 8 * hf + r) * C_ + h * DH_ + j * 16 + ml] =
          acc[j][r] * inv;
  }
}

// ---------------------------------------------------------------------------
extern "C" void kernel_launch(void* const* d_in, const int* in_sizes, int n_in,
                              void* d_out, int out_size, void* d_ws, size_t ws_size,
                              hipStream_t stream) {
  (void)in_sizes; (void)n_in; (void)out_size; (void)ws_size;
  const float* x     = (const float*)d_in[0];
  const float* text  = (const float*)d_in[1];
  const float* sigma = (const float*)d_in[2];
  const float* pe    = (const float*)d_in[4];
  const float* td_w  = (const float*)d_in[5];
  const float* td_b  = (const float*)d_in[6];
  const float* wq = (const float*)d_in[7];   const float* bq = (const float*)d_in[8];
  const float* wk = (const float*)d_in[9];   const float* bk = (const float*)d_in[10];
  const float* wv = (const float*)d_in[11];  const float* bv = (const float*)d_in[12];
  const float* wo = (const float*)d_in[13];  const float* bo = (const float*)d_in[14];
  const float* fw1 = (const float*)d_in[15]; const float* fb1 = (const float*)d_in[16];
  const float* fw2 = (const float*)d_in[17]; const float* fb2 = (const float*)d_in[18];
  const float* afw[16];
  for (int i = 0; i < 16; ++i) afw[i] = (const float*)d_in[19 + i];

  float* out = (float*)d_out;
  float* att = out + (size_t)BS_ * C_;   // att output region (also softmax scratch)

  char* wsp = (char*)d_ws;
  auto alloc = [&](size_t bytes) {
    void* p = (void*)wsp;
    wsp += (bytes + 255) & ~(size_t)255;
    return p;
  };
  float* text1  = (float*)alloc((size_t)BT_ * C_ * 4);
  float* textpe = (float*)alloc((size_t)BT_ * C_ * 4);
  float* qf = (float*)alloc((size_t)BS_ * C_ * 4);
  float* kf = (float*)alloc((size_t)BS_ * C_ * 4);
  float* vf = (float*)alloc((size_t)BS_ * C_ * 4);
  float* of = (float*)alloc((size_t)BS_ * C_ * 4);
  float* x2 = (float*)alloc((size_t)BS_ * C_ * 4);
  float* x3 = (float*)alloc((size_t)BS_ * C_ * 4);
  float* hb = (float*)alloc((size_t)BS_ * 2 * C_ * 4);
  unsigned* qbb = (unsigned*)alloc((size_t)BS_ * (C_ / 2) * 4);
  unsigned* kbb = (unsigned*)alloc((size_t)BS_ * (C_ / 2) * 4);
  unsigned* vtb = (unsigned*)alloc((size_t)B_ * H_ * DH_ * (S_ / 2) * 4);
  float* gbuf = (float*)alloc((size_t)8 * B_ * C_ * 4);
  const size_t BC = (size_t)B_ * C_;

  // FiLM gammas/betas (slots: g0,b0,g1,b1,g2,b2,g3,b3)
  for (int i = 0; i < 4; ++i) {
    film_kernel<<<(B_ * C_) / 256, 256, 0, stream>>>(sigma, afw[4 * i + 0], afw[4 * i + 1],
                                                     gbuf + (2 * i + 0) * BC);
    film_kernel<<<(B_ * C_) / 256, 256, 0, stream>>>(sigma, afw[4 * i + 2], afw[4 * i + 3],
                                                     gbuf + (2 * i + 1) * BC);
  }

  // text1 = swish(text) @ td_w + td_b
  gemm_bf16_kernel<<<dim3(C_ / 128, BT_ / 128), 256, 0, stream>>>(
      text, td_w, td_b, nullptr, text1, BT_, C_, C_, 1, 0);
  // text1 = affine(LN(text1)) ; textpe = text1 + pe[:T]
  ln_affine_kernel<<<BT_ / 8, 256, 0, stream>>>(text1, nullptr, gbuf + 0 * BC,
                                                gbuf + 1 * BC, nullptr, pe, text1, textpe, T_);
  // MHA1 projections
  gemm_bf16_kernel<<<dim3(C_ / 128, BS_ / 128), 256, 0, stream>>>(
      x, wq, bq, nullptr, qf, BS_, C_, C_, 0, 0);
  gemm_bf16_kernel<<<dim3(C_ / 128, BT_ / 128), 256, 0, stream>>>(
      textpe, wk, bk, nullptr, kf, BT_, C_, C_, 0, 0);
  gemm_bf16_kernel<<<dim3(C_ / 128, BT_ / 128), 256, 0, stream>>>(
      text1, wv, bv, nullptr, vf, BT_, C_, C_, 0, 0);
  conv_bf16_kernel<<<(BS_ * C_ / 2) / 256, 256, 0, stream>>>(qf, qbb, BS_ * C_ / 2, 0.125f);
  conv_bf16_kernel<<<(BT_ * C_ / 2) / 256, 256, 0, stream>>>(kf, kbb, BT_ * C_ / 2, 1.0f);
  convT_v_kernel<<<(B_ * H_ * DH_ * T_ / 2) / 256, 256, 0, stream>>>(vf, vtb, T_);
  // scores -> softmax (att is a model output) -> att @ V
  mha1_scores_kernel<<<B_ * H_ * (S_ / 16), 32, 0, stream>>>(qbb, kbb, att);
  softmax256_kernel<<<(B_ * H_ * S_) / 8, 256, 0, stream>>>(att);
  mha1_attv_kernel<<<B_ * H_ * (S_ / 16), 32, 0, stream>>>(att, vtb, of);
  // x2 = affine(LN(o @ wo + bo)) + x
  gemm_bf16_kernel<<<dim3(C_ / 128, BS_ / 128), 256, 0, stream>>>(
      of, wo, bo, nullptr, x2, BS_, C_, C_, 0, 0);
  ln_affine_kernel<<<BS_ / 8, 256, 0, stream>>>(x2, nullptr, gbuf + 2 * BC, gbuf + 3 * BC,
                                                x, nullptr, x2, nullptr, S_);
  // MHA2 (self) projections from x2
  gemm_bf16_kernel<<<dim3(C_ / 128, BS_ / 128), 256, 0, stream>>>(
      x2, wq, bq, nullptr, qf, BS_, C_, C_, 0, 0);
  gemm_bf16_kernel<<<dim3(C_ / 128, BS_ / 128), 256, 0, stream>>>(
      x2, wk, bk, nullptr, kf, BS_, C_, C_, 0, 0);
  gemm_bf16_kernel<<<dim3(C_ / 128, BS_ / 128), 256, 0, stream>>>(
      x2, wv, bv, nullptr, vf, BS_, C_, C_, 0, 0);
  conv_bf16_kernel<<<(BS_ * C_ / 2) / 256, 256, 0, stream>>>(qf, qbb, BS_ * C_ / 2, 0.125f);
  conv_bf16_kernel<<<(BS_ * C_ / 2) / 256, 256, 0, stream>>>(kf, kbb, BS_ * C_ / 2, 1.0f);
  convT_v_kernel<<<(B_ * H_ * DH_ * S_ / 2) / 256, 256, 0, stream>>>(vf, vtb, S_);
  flash_kernel<<<B_ * H_ * (S_ / 16), 32, 0, stream>>>(qbb, kbb, vtb, of);
  // x3 = affine(LN(x2 + o @ wo + bo))
  gemm_bf16_kernel<<<dim3(C_ / 128, BS_ / 128), 256, 0, stream>>>(
      of, wo, bo, nullptr, x3, BS_, C_, C_, 0, 0);
  ln_affine_kernel<<<BS_ / 8, 256, 0, stream>>>(x3, x2, gbuf + 4 * BC, gbuf + 5 * BC,
                                                nullptr, nullptr, x3, nullptr, S_);
  // FFN: h = swish(swish(x3) @ W1 + b1); x4 = h @ W2 + b2 + x3 (reuse qf)
  gemm_bf16_kernel<<<dim3(2 * C_ / 128, BS_ / 128), 256, 0, stream>>>(
      x3, fw1, fb1, nullptr, hb, BS_, 2 * C_, C_, 1, 1);
  gemm_bf16_kernel<<<dim3(C_ / 128, BS_ / 128), 256, 0, stream>>>(
      hb, fw2, fb2, x3, qf, BS_, C_, 2 * C_, 0, 0);
  // out = affine(LN(x4))
  ln_affine_kernel<<<BS_ / 8, 256, 0, stream>>>(qf, nullptr, gbuf + 6 * BC, gbuf + 7 * BC,
                                                nullptr, nullptr, out, nullptr, S_);
}
